// GroupedQueryAttention_66571993088382
// MI455X (gfx1250) — matrix operand order repaired and checked by reading, not compiled, not run on hardware
//
#include <hip/hip_runtime.h>
#include <stdint.h>

typedef __attribute__((ext_vector_type(16))) __bf16 v16bf;
typedef __attribute__((ext_vector_type(8)))  float  v8f;
typedef __attribute__((ext_vector_type(4)))  float  v4f;

#define DEVINL __device__ __forceinline__

// ---------------------------------------------------------------------------
// WMMA helpers (CDNA5 v_wmma_f32_16x16x32_bf16, wave32)
// ---------------------------------------------------------------------------
DEVINL v8f wmma_bf16f32(v16bf a, v16bf b, v8f c) {
  // (neg_a, A, neg_b, B, c_mod, C, reuse_a, reuse_b)
  return __builtin_amdgcn_wmma_f32_16x16x32_bf16(false, a, false, b, (short)0, c,
                                                 false, false);
}

// ---------------------------------------------------------------------------
// CDNA5 async global->LDS copy (ASYNCcnt-tracked, bypasses VGPRs).
// Each active lane copies 16 bytes: LDS[lds_dst] = MEM[gsrc .. gsrc+15].
// ---------------------------------------------------------------------------
DEVINL void async_copy_b128(const void* gsrc, void* lds_dst) {
  const unsigned lds = (unsigned)(uintptr_t)lds_dst;  // low 32b of generic = LDS offset
  asm volatile("global_load_async_to_lds_b128 %0, %1, off"
               :
               : "v"(lds), "v"(gsrc)
               : "memory");
}
DEVINL void wait_async0() { asm volatile("s_wait_asynccnt 0x0" ::: "memory"); }

// A-matrix 16x32 bf16 layout (contiguous K, matching B fragments):
// lanes 0-15: M=lane, VGPR v -> K=2v,2v+1 (K=0..15)
// lanes 16-31: same but K base +16.
DEVINL int frag_ka(int v, int kb) { return kb + 2 * v; }

DEVINL v16bf load_frag_a(const __bf16* base, int stride) {
  const int lane = threadIdx.x & 31;
  const int row  = lane & 15;
  const int kb   = (lane >> 4) << 4;  // 0 or 16
  const __bf16* rp = base + (size_t)row * stride;
  v16bf f;
#pragma unroll
  for (int v = 0; v < 8; ++v) {
    const int kk = frag_ka(v, kb);
    f[2 * v]     = rp[kk];
    f[2 * v + 1] = rp[kk + 1];
  }
  return f;
}

// B-matrix 32x16 bf16: lane = column; lanes 0-15 hold K=0..15, lanes 16-31 K=16..31.
// Element B(k,n) = base[k*stride + n].
DEVINL v16bf load_frag_b(const __bf16* base, int stride) {
  const int lane = threadIdx.x & 31;
  const int col  = lane & 15;
  const int kb   = (lane >> 4) << 4;  // 0 or 16
  v16bf f;
#pragma unroll
  for (int v = 0; v < 8; ++v) {
    const int kk = kb + 2 * v;
    f[2 * v]     = base[(size_t)kk * stride + col];
    f[2 * v + 1] = base[(size_t)(kk + 1) * stride + col];
  }
  return f;
}

// Same fragment but B(k,n) = base[n*stride + k]  (reads a row-major K-block as K^T)
DEVINL v16bf load_frag_bT(const __bf16* base, int stride) {
  const int lane = threadIdx.x & 31;
  const int col  = lane & 15;
  const int kb   = (lane >> 4) << 4;
  const __bf16* cp = base + (size_t)col * stride;
  v16bf f;
#pragma unroll
  for (int v = 0; v < 8; ++v) {
    const int kk = kb + 2 * v;
    f[2 * v]     = cp[kk];
    f[2 * v + 1] = cp[kk + 1];
  }
  return f;
}

// ---------------------------------------------------------------------------
// f32 -> bf16 convert
// ---------------------------------------------------------------------------
__global__ void f32_to_bf16(const float* __restrict__ src, __bf16* __restrict__ dst,
                            int n) {
  int i = blockIdx.x * blockDim.x + threadIdx.x;
  const int stride = gridDim.x * blockDim.x;
  for (; i < n; i += stride) dst[i] = (__bf16)src[i];
}

// ---------------------------------------------------------------------------
// bf16 GEMM: C[MxN] = A[MxK] * B[KxN], fp32 accumulate. 64x64 tile / block,
// 4 waves (128 thr), each wave a 32x32 subtile (2x2 WMMA frags), BK=32,
// double-buffered LDS fed by async global->LDS DMA.
// ---------------------------------------------------------------------------
template <bool STORE_BF16>
__global__ __launch_bounds__(128) void gemm_bf16(const __bf16* __restrict__ A,
                                                 const __bf16* __restrict__ Bm,
                                                 void* __restrict__ C, int M, int N,
                                                 int K) {
  constexpr int BM = 64, BN = 64, BK = 32;
  __shared__ alignas(16) __bf16 As[2][BM][BK + 8];
  __shared__ alignas(16) __bf16 Bs[2][BK][BN + 8];
  const int tid  = threadIdx.x;
  const int wave = tid >> 5;
  const int lane = tid & 31;
  const int lo = lane & 15, hi = lane >> 4;
  const int m0 = blockIdx.y * BM;
  const int n0 = blockIdx.x * BN;
  const int wm = (wave >> 1) * 32;
  const int wn = (wave & 1) * 32;

  v8f acc[2][2] = {};

  auto stage = [&](int buf, int kb) {
#pragma unroll
    for (int c = 0; c < 2; ++c) {  // A tile 64x32 = 2048 elems, 8 per thread x2
      const int e = (tid + c * 128) * 8;
      const int r = e >> 5, col = e & 31;
      async_copy_b128(A + (size_t)(m0 + r) * K + kb + col, &As[buf][r][col]);
    }
#pragma unroll
    for (int c = 0; c < 2; ++c) {  // B tile 32x64
      const int e = (tid + c * 128) * 8;
      const int r = e >> 6, col = e & 63;
      async_copy_b128(Bm + (size_t)(kb + r) * N + n0 + col, &Bs[buf][r][col]);
    }
  };

  const int nk = K / BK;
  stage(0, 0);
  wait_async0();
  __syncthreads();
  for (int k = 0; k < nk; ++k) {
    const int buf = k & 1;
    if (k + 1 < nk) stage(buf ^ 1, (k + 1) * BK);  // async DMA overlaps WMMAs below
    if (k + 2 < nk)                                // global_prefetch_b8 of tile k+2
      __builtin_prefetch(A + (size_t)(m0 + (tid >> 1)) * K + (k + 2) * BK + (tid & 1) * 16,
                         0, 0);
    v16bf a0 = load_frag_a(&As[buf][wm + 0][0], BK + 8);
    v16bf a1 = load_frag_a(&As[buf][wm + 16][0], BK + 8);
    v16bf b0 = load_frag_b(&Bs[buf][0][wn + 0], BN + 8);
    v16bf b1 = load_frag_b(&Bs[buf][0][wn + 16], BN + 8);
    acc[0][0] = wmma_bf16f32(a0, b0, acc[0][0]);
    acc[0][1] = wmma_bf16f32(a0, b1, acc[0][1]);
    acc[1][0] = wmma_bf16f32(a1, b0, acc[1][0]);
    acc[1][1] = wmma_bf16f32(a1, b1, acc[1][1]);
    wait_async0();
    __syncthreads();
  }

#pragma unroll
  for (int i = 0; i < 2; ++i)
#pragma unroll
    for (int j = 0; j < 2; ++j) {
      const int r0 = m0 + wm + i * 16 + hi * 8;
      const int cc = n0 + wn + j * 16 + lo;
#pragma unroll
      for (int vv = 0; vv < 8; ++vv) {
        const float val = acc[i][j][vv];
        if (STORE_BF16)
          ((__bf16*)C)[(size_t)(r0 + vv) * N + cc] = (__bf16)val;
        else
          ((float*)C)[(size_t)(r0 + vv) * N + cc] = val;
      }
    }
}

// ---------------------------------------------------------------------------
// In-place RoPE on bf16 [B*T, heads*64]; one thread per (bt, h, pair)
// ---------------------------------------------------------------------------
__global__ void rope_inplace(__bf16* __restrict__ qk, const float* __restrict__ cs,
                             const float* __restrict__ sn, int heads, int npairs) {
  const int i = blockIdx.x * blockDim.x + threadIdx.x;
  if (i >= npairs) return;
  const int p  = i & 31;               // HEAD_DIM/2 = 32 pairs
  const int h  = (i >> 5) % heads;
  const int bt = i / (32 * heads);
  const int t  = bt & 2047;            // T = 2048
  const float c = cs[t * 32 + p], s = sn[t * 32 + p];
  const size_t base = (size_t)bt * heads * 64 + h * 64 + 2 * p;
  const float xr = (float)qk[base];
  const float xi = (float)qk[base + 1];
  qk[base]     = (__bf16)(xr * c - xi * s);
  qk[base + 1] = (__bf16)(xr * s + xi * c);
}

// ---------------------------------------------------------------------------
// Flash attention (causal, GQA). Block = 128 thr (4 waves) handles 64 query
// rows for one (b, h). Each wave owns 16 rows. K/V blocks of 64 keys are
// double-buffered in LDS via async DMA; S = Q K^T and O += P V on WMMA;
// online softmax in fp32.
// ---------------------------------------------------------------------------
__global__ __launch_bounds__(128) void flash_attn(const __bf16* __restrict__ q,
                                                  const __bf16* __restrict__ k,
                                                  const __bf16* __restrict__ v,
                                                  __bf16* __restrict__ out) {
  constexpr int T = 2048, HQ = 16, HKV = 4, HD = 64;
  const int qt  = blockIdx.x;   // query tile (64 rows)
  const int h   = blockIdx.y;
  const int b   = blockIdx.z;
  const int kvh = h >> 2;       // N_REP = 4
  const int q0  = qt * 64;
  const int tid = threadIdx.x, wave = tid >> 5, lane = tid & 31;
  const int lo = lane & 15, hi = lane >> 4;

  __shared__ alignas(16) __bf16 Ks[2][64][HD + 8];
  __shared__ alignas(16) __bf16 Vs[2][64][HD + 8];
  __shared__ alignas(16) __bf16 Ps[4][16][64 + 8];  // per-wave P refragment scratch

  auto stage = [&](int buf, int kb) {
#pragma unroll
    for (int c = 0; c < 4; ++c) {
      const int e = (tid + c * 128) * 8;
      const int r = e >> 6, col = e & 63;
      const size_t row = (size_t)(b * T + kb + r) * (HKV * HD) + kvh * HD + col;
      async_copy_b128(k + row, &Ks[buf][r][col]);
      async_copy_b128(v + row, &Vs[buf][r][col]);
    }
  };

  // Q A-fragments, held in registers for the whole kv loop
  const __bf16* qbase = q + (size_t)(b * T + q0 + wave * 16) * (HQ * HD) + h * HD;
  const v16bf aq0 = load_frag_a(qbase, HQ * HD);       // d = 0..31
  const v16bf aq1 = load_frag_a(qbase + 32, HQ * HD);  // d = 32..63

  v8f o[4] = {};
  float mrow[8], lrow[8];
#pragma unroll
  for (int vv = 0; vv < 8; ++vv) { mrow[vv] = -3.0e38f; lrow[vv] = 0.0f; }
  const float scale = 0.125f;  // 1/sqrt(64)

  stage(0, 0);
  wait_async0();
  __syncthreads();

  for (int j = 0; j <= qt; ++j) {
    const int kb  = j * 64;
    const int buf = j & 1;
    if (j < qt) stage(buf ^ 1, kb + 64);  // async DMA for next block overlaps compute

    // S = Q K^T  (16 x 64 per wave)
    v8f s[4] = {};
#pragma unroll
    for (int nt = 0; nt < 4; ++nt) {
      v16bf bk0 = load_frag_bT(&Ks[buf][nt * 16][0], HD + 8);
      s[nt] = wmma_bf16f32(aq0, bk0, s[nt]);
      v16bf bk1 = load_frag_bT(&Ks[buf][nt * 16][32], HD + 8);
      s[nt] = wmma_bf16f32(aq1, bk1, s[nt]);
    }

    // scale + causal mask (mask only fires on the diagonal block)
    const int rbase = q0 + wave * 16 + hi * 8;
#pragma unroll
    for (int nt = 0; nt < 4; ++nt) {
      const int colg = kb + nt * 16 + lo;
#pragma unroll
      for (int vv = 0; vv < 8; ++vv) {
        const float xv = s[nt][vv] * scale;
        s[nt][vv] = (colg > rbase + vv) ? -3.0e38f : xv;
      }
    }

    // online softmax; row elements live across the 16 lanes of a half-wave
#pragma unroll
    for (int vv = 0; vv < 8; ++vv) {
      float mx = fmaxf(fmaxf(s[0][vv], s[1][vv]), fmaxf(s[2][vv], s[3][vv]));
#pragma unroll
      for (int off = 1; off < 16; off <<= 1) mx = fmaxf(mx, __shfl_xor(mx, off, 32));
      const float mnew = fmaxf(mrow[vv], mx);
      const float corr = __expf(mrow[vv] - mnew);
      mrow[vv] = mnew;
      float rs = 0.0f;
#pragma unroll
      for (int nt = 0; nt < 4; ++nt) {
        const float p = __expf(s[nt][vv] - mnew);
        s[nt][vv] = p;
        rs += p;
      }
#pragma unroll
      for (int off = 1; off < 16; off <<= 1) rs += __shfl_xor(rs, off, 32);
      lrow[vv] = lrow[vv] * corr + rs;
#pragma unroll
      for (int nt = 0; nt < 4; ++nt) o[nt][vv] *= corr;
    }

    // refragment P (C-layout -> A-layout) through per-wave LDS scratch
#pragma unroll
    for (int nt = 0; nt < 4; ++nt)
#pragma unroll
      for (int vv = 0; vv < 8; ++vv)
        Ps[wave][hi * 8 + vv][nt * 16 + lo] = (__bf16)s[nt][vv];
    asm volatile("s_wait_dscnt 0x0" ::: "memory");  // own-wave DS stores visible

    // O += P V
#pragma unroll
    for (int ks2 = 0; ks2 < 2; ++ks2) {
      v16bf ap = load_frag_a(&Ps[wave][0][ks2 * 32], 64 + 8);
#pragma unroll
      for (int nt = 0; nt < 4; ++nt) {
        v16bf bv = load_frag_b(&Vs[buf][ks2 * 32][nt * 16], HD + 8);
        o[nt] = wmma_bf16f32(ap, bv, o[nt]);
      }
    }
    wait_async0();
    __syncthreads();
  }

  // normalize and store to [b, t, h*64 + d] (bf16)
#pragma unroll
  for (int vv = 0; vv < 8; ++vv) {
    const float inv = 1.0f / lrow[vv];
    const int t = q0 + wave * 16 + hi * 8 + vv;
    const size_t rb = (size_t)(b * T + t) * (HQ * HD) + h * HD;
#pragma unroll
    for (int nt = 0; nt < 4; ++nt)
      out[rb + nt * 16 + lo] = (__bf16)(o[nt][vv] * inv);
  }
}

// ---------------------------------------------------------------------------
// Host-side orchestration
// ---------------------------------------------------------------------------
extern "C" void kernel_launch(void* const* d_in, const int* in_sizes, int n_in,
                              void* d_out, int out_size, void* d_ws, size_t ws_size,
                              hipStream_t stream) {
  (void)in_sizes; (void)n_in; (void)out_size; (void)ws_size;
  constexpr int Bc = 2, Tc = 2048, Dm = 1024, HQ = 16, HKV = 4, HD = 64;
  constexpr int M = Bc * Tc;  // 4096

  const float* x  = (const float*)d_in[0];
  const float* fc = (const float*)d_in[1];
  const float* fs = (const float*)d_in[2];
  const float* wq = (const float*)d_in[3];
  const float* wk = (const float*)d_in[4];
  const float* wv = (const float*)d_in[5];
  const float* wo = (const float*)d_in[6];
  float* out = (float*)d_out;

  char* ws = (char*)d_ws;
  size_t off = 0;
  auto alloc = [&](size_t elems) -> __bf16* {
    off = (off + 255) & ~(size_t)255;
    __bf16* p = (__bf16*)(ws + off);
    off += elems * sizeof(__bf16);
    return p;
  };
  __bf16* xb   = alloc((size_t)M * Dm);
  __bf16* wqb  = alloc((size_t)Dm * Dm);
  __bf16* wkb  = alloc((size_t)Dm * (HKV * HD));
  __bf16* wvb  = alloc((size_t)Dm * (HKV * HD));
  __bf16* wob  = alloc((size_t)Dm * Dm);
  __bf16* qlin = alloc((size_t)M * Dm);
  __bf16* klin = alloc((size_t)M * (HKV * HD));
  __bf16* vlin = alloc((size_t)M * (HKV * HD));
  __bf16* attn = alloc((size_t)M * Dm);

  auto cvt = [&](const float* s, __bf16* d, int n) {
    int blocks = (n + 255) / 256;
    if (blocks > 4096) blocks = 4096;
    f32_to_bf16<<<blocks, 256, 0, stream>>>(s, d, n);
  };
  cvt(x, xb, M * Dm);
  cvt(wq, wqb, Dm * Dm);
  cvt(wk, wkb, Dm * HKV * HD);
  cvt(wv, wvb, Dm * HKV * HD);
  cvt(wo, wob, Dm * Dm);

  // QKV projections (bf16 WMMA, fp32 accum, bf16 out)
  gemm_bf16<true><<<dim3(Dm / 64, M / 64), 128, 0, stream>>>(xb, wqb, qlin, M, Dm, Dm);
  gemm_bf16<true><<<dim3((HKV * HD) / 64, M / 64), 128, 0, stream>>>(xb, wkb, klin, M,
                                                                     HKV * HD, Dm);
  gemm_bf16<true><<<dim3((HKV * HD) / 64, M / 64), 128, 0, stream>>>(xb, wvb, vlin, M,
                                                                     HKV * HD, Dm);

  // RoPE (in place on freshly computed Q/K)
  {
    int np = M * HQ * 32;
    rope_inplace<<<(np + 255) / 256, 256, 0, stream>>>(qlin, fc, fs, HQ, np);
    np = M * HKV * 32;
    rope_inplace<<<(np + 255) / 256, 256, 0, stream>>>(klin, fc, fs, HKV, np);
  }

  // Causal GQA flash attention
  flash_attn<<<dim3(Tc / 64, HQ, Bc), 128, 0, stream>>>(qlin, klin, vlin, attn);

  // Output projection, fp32 result
  gemm_bf16<false><<<dim3(Dm / 64, M / 64), 128, 0, stream>>>(attn, wob, out, M, Dm, Dm);
}
